// Net_3951369912443
// MI455X (gfx1250) — compile-verified, matrix operand order checked by reading, not run on hardware
//
#include <hip/hip_runtime.h>
#include <hip/hip_bf16.h>

// ---------------------------------------------------------------------------
// APPNP: log_softmax( PPR_K( relu(X W1 + b1) W2 + b2 ) ) on gfx1250 (MI455X)
//   N=100000 nodes, E=1600000 edges, F=512, H=128, C=64, K=10, alpha=0.1
//
// GEMMs: v_wmma_f32_16x16x32_bf16 (wave32 WMMA).
//   - A operand staged through LDS (coalesced loads, one-time f32->bf16).
//   - B operand pre-packed into the exact WMMA per-lane fragment layout, so
//     the inner loop is 2x ds_load_b128 + 2x global_load_b128 + 1x v_wmma.
// Propagation: bandwidth/atomic bound; coalesced float2 gathers + f32 atomics
// that resolve in L2 (25.6 MB accumulator << 192 MB L2).
// ---------------------------------------------------------------------------

typedef __attribute__((ext_vector_type(16))) __bf16 v16bf;
typedef __attribute__((ext_vector_type(8)))  __bf16 v8bf;
typedef __attribute__((ext_vector_type(8)))  float  v8f;

#define NFEAT 512
#define NHID  128
#define NCLS  64

// ---- degree / normalization -----------------------------------------------
__global__ __launch_bounds__(256) void init_deg(float* deg, int n) {
    int i = blockIdx.x * 256 + threadIdx.x;
    if (i < n) deg[i] = 1.0f;                 // self-loop
}

__global__ __launch_bounds__(256) void accum_deg(const int* __restrict__ dst,
                                                 float* __restrict__ deg, int e) {
    int i = blockIdx.x * 256 + threadIdx.x;
    if (i < e) atomicAdd(&deg[dst[i]], 1.0f);
}

__global__ __launch_bounds__(256) void finish_dinv(float* dinv, int n) {
    int i = blockIdx.x * 256 + threadIdx.x;
    if (i < n) dinv[i] = rsqrtf(dinv[i]);     // deg >= 1 always (self-loop)
}

// ---- pack W[K][Ncols] (f32) into WMMA B-fragment layout (bf16) ------------
// out element t=((colTile*ksteps + ks)*32 + lane), 16 bf16 per t:
//   e<8 : K = ks*32 + khalf + e        (khalf = (lane>=16) ? 8 : 0)
//   e>=8: K = ks*32 + khalf + 16 + e-8 , col = colTile*16 + (lane&15)
__global__ __launch_bounds__(128) void pack_B(const float* __restrict__ W,
                                              __bf16* __restrict__ out,
                                              int Kdim, int Ncols) {
    int ksteps = Kdim >> 5;
    int total = (Ncols >> 4) * ksteps * 32;
    int t = blockIdx.x * 128 + threadIdx.x;
    if (t >= total) return;
    int lane = t & 31;
    int ks = (t >> 5) % ksteps;
    int ct = (t >> 5) / ksteps;
    int col = (ct << 4) + (lane & 15);
    int kbase = (ks << 5) + ((lane >> 4) << 3);
    __bf16* o = out + (size_t)t * 16;
#pragma unroll
    for (int e = 0; e < 8; ++e) {
        o[e]     = (__bf16)W[(size_t)(kbase + e) * Ncols + col];
        o[8 + e] = (__bf16)W[(size_t)(kbase + 16 + e) * Ncols + col];
    }
}

// ---- GEMM1: H1 = relu(X[N,512] @ W1[512,128] + b1) -> bf16 [N,128] --------
// grid.x = N/16 row tiles; 256 threads = 8 waves; wave w owns col tile w*16.
__global__ __launch_bounds__(256) void gemm1_relu_wmma(
    const float* __restrict__ X, const __bf16* __restrict__ W1pk,
    const float* __restrict__ b1, __bf16* __restrict__ H1) {
    __shared__ __bf16 lx[16][NFEAT + 8];   // +8 pad -> conflict-free b128 reads
    const int rowTile = blockIdx.x;

    // cooperative, coalesced X-tile load with one-time f32->bf16 conversion
    const float* xt = X + (size_t)rowTile * 16 * NFEAT;
    for (int f = threadIdx.x; f < 16 * (NFEAT / 4); f += 256) {
        int r = f >> 7;                    // NFEAT/4 = 128 float4 per row
        int c4 = (f & 127) << 2;
        float4 v = *(const float4*)(xt + (size_t)r * NFEAT + c4);
        lx[r][c4 + 0] = (__bf16)v.x;
        lx[r][c4 + 1] = (__bf16)v.y;
        lx[r][c4 + 2] = (__bf16)v.z;
        lx[r][c4 + 3] = (__bf16)v.w;
    }
    __syncthreads();

    const int wave  = threadIdx.x >> 5;
    const int lane  = threadIdx.x & 31;
    const int col   = (wave << 4) + (lane & 15);
    const int arow  = lane & 15;
    const int khalf = (lane >> 4) << 3;    // 0 or 8
    const __bf16* bp = W1pk + ((size_t)wave * (NFEAT / 32) * 32 + lane) * 16;

    v8f c = {};
#pragma unroll 4
    for (int ks = 0; ks < NFEAT / 32; ++ks) {
        const __bf16* ar = &lx[arow][(ks << 5) + khalf];
        v8bf alo = *(const v8bf*)(ar);         // K = 32ks+khalf+0..7
        v8bf ahi = *(const v8bf*)(ar + 16);    // K = 32ks+khalf+16..23
        const v8bf* bq = (const v8bf*)(bp + (size_t)ks * 32 * 16);
        v8bf blo = bq[0];
        v8bf bhi = bq[1];
        v16bf a, b;
#pragma unroll
        for (int e = 0; e < 8; ++e) {
            a[e] = alo[e]; a[8 + e] = ahi[e];
            b[e] = blo[e]; b[8 + e] = bhi[e];
        }
        c = __builtin_amdgcn_wmma_f32_16x16x32_bf16(false, a, false, b,
                                                    (short)0, c, false, false);
    }
    const float bias = b1[col];
    const int mbase = (lane >> 4) << 3;
#pragma unroll
    for (int v = 0; v < 8; ++v) {
        float r = c[v] + bias;
        r = r > 0.0f ? r : 0.0f;               // ReLU
        H1[(size_t)((rowTile << 4) + v + mbase) * NHID + col] = (__bf16)r;
    }
}

// ---- GEMM2: H = H1[N,128]bf16 @ W2[128,64] + b2 -> x0, hcur (f32) ---------
// grid.x = N/16; 128 threads = 4 waves; wave w owns col tile w*16 (C=64).
__global__ __launch_bounds__(128) void gemm2_wmma(
    const __bf16* __restrict__ H1, const __bf16* __restrict__ W2pk,
    const float* __restrict__ b2, float* __restrict__ x0,
    float* __restrict__ hcur) {
    __shared__ __bf16 lh[16][NHID + 8];
    const int rowTile = blockIdx.x;

    // cooperative tile load (bf16 already): 16 rows x 128 = 256 x 16B chunks
    const __bf16* ht = H1 + (size_t)rowTile * 16 * NHID;
    for (int ch = threadIdx.x; ch < 16 * (NHID / 8); ch += 128) {
        int r = ch >> 4;                   // 16 chunks of 8 bf16 per row
        int off = (ch & 15) << 3;
        *(v8bf*)(&lh[r][off]) = *(const v8bf*)(ht + (size_t)r * NHID + off);
    }
    __syncthreads();

    const int wave  = threadIdx.x >> 5;
    const int lane  = threadIdx.x & 31;
    const int col   = (wave << 4) + (lane & 15);
    const int arow  = lane & 15;
    const int khalf = (lane >> 4) << 3;
    const __bf16* bp = W2pk + ((size_t)wave * (NHID / 32) * 32 + lane) * 16;

    v8f c = {};
#pragma unroll
    for (int ks = 0; ks < NHID / 32; ++ks) {
        const __bf16* ar = &lh[arow][(ks << 5) + khalf];
        v8bf alo = *(const v8bf*)(ar);
        v8bf ahi = *(const v8bf*)(ar + 16);
        const v8bf* bq = (const v8bf*)(bp + (size_t)ks * 32 * 16);
        v8bf blo = bq[0];
        v8bf bhi = bq[1];
        v16bf a, b;
#pragma unroll
        for (int e = 0; e < 8; ++e) {
            a[e] = alo[e]; a[8 + e] = ahi[e];
            b[e] = blo[e]; b[8 + e] = bhi[e];
        }
        c = __builtin_amdgcn_wmma_f32_16x16x32_bf16(false, a, false, b,
                                                    (short)0, c, false, false);
    }
    const float bias = b2[col];
    const int mbase = (lane >> 4) << 3;
#pragma unroll
    for (int v = 0; v < 8; ++v) {
        float r = c[v] + bias;
        size_t idx = (size_t)((rowTile << 4) + v + mbase) * NCLS + col;
        x0[idx] = r;
        hcur[idx] = r;
    }
}

// ---- propagation: one wave32 per edge, lane covers 2 consecutive classes --
__global__ __launch_bounds__(256) void edge_scatter(
    const int* __restrict__ src, const int* __restrict__ dst,
    const float* __restrict__ dinv, const float* __restrict__ hcur,
    float* __restrict__ hagg, int E) {
    int e = blockIdx.x * 8 + (threadIdx.x >> 5);
    if (e >= E) return;
    int lane = threadIdx.x & 31;
    int s = src[e];
    int d = dst[e];
    float w = dinv[s] * dinv[d];                       // symmetric GCN norm
    float2 hv = *(const float2*)(hcur + (size_t)s * NCLS + lane * 2);
    float* out = hagg + (size_t)d * NCLS + lane * 2;
    atomicAdd(out,     w * hv.x);                      // global_atomic_add_f32
    atomicAdd(out + 1, w * hv.y);
}

// ---- h = (1-a)*(agg + dinv[i]^2 * h) + a * x0  (self-loop folded in) ------
__global__ __launch_bounds__(256) void combine(
    const float* __restrict__ hagg, const float* __restrict__ x0,
    const float* __restrict__ dinv, float* __restrict__ hcur, int total) {
    int t = blockIdx.x * 256 + threadIdx.x;
    if (t >= total) return;
    int i = t >> 6;
    float di = dinv[i];
    float self = di * di * hcur[t];
    hcur[t] = 0.9f * (hagg[t] + self) + 0.1f * x0[t];
}

// ---- log_softmax over 64 classes: one wave per node, 2 classes per lane ---
__global__ __launch_bounds__(256) void log_softmax64(
    const float* __restrict__ h, float* __restrict__ out, int N) {
    int node = blockIdx.x * 8 + (threadIdx.x >> 5);
    if (node >= N) return;
    int lane = threadIdx.x & 31;
    float v0 = h[(size_t)node * NCLS + lane];
    float v1 = h[(size_t)node * NCLS + 32 + lane];
    float m = fmaxf(v0, v1);
#pragma unroll
    for (int o = 16; o > 0; o >>= 1) m = fmaxf(m, __shfl_xor(m, o, 32));
    float s = __expf(v0 - m) + __expf(v1 - m);
#pragma unroll
    for (int o = 16; o > 0; o >>= 1) s += __shfl_xor(s, o, 32);
    float lse = m + __logf(s);
    out[(size_t)node * NCLS + lane]      = v0 - lse;
    out[(size_t)node * NCLS + 32 + lane] = v1 - lse;
}

// ---------------------------------------------------------------------------
extern "C" void kernel_launch(void* const* d_in, const int* in_sizes, int n_in,
                              void* d_out, int out_size, void* d_ws, size_t ws_size,
                              hipStream_t stream) {
    const float* X     = (const float*)d_in[0];
    const int*   edges = (const int*)d_in[1];   // [2,E]: row 0 = src, row 1 = dst
    const float* W1    = (const float*)d_in[2];
    const float* b1    = (const float*)d_in[3];
    const float* W2    = (const float*)d_in[4];
    const float* b2    = (const float*)d_in[5];
    float* out = (float*)d_out;

    const int N = in_sizes[0] / NFEAT;          // 100000 (multiple of 16)
    const int E = in_sizes[1] / 2;              // 1600000
    const int* src = edges;
    const int* dst = edges + E;

    // workspace layout (16B-aligned slices)
    const size_t NC = (size_t)N * NCLS;
    float*  dinv = (float*)d_ws;                // N floats (deg -> dinv in place)
    float*  x0   = dinv + 102400;               // NC floats
    float*  hcur = x0 + NC;                     // NC floats
    float*  hagg = hcur + NC;                   // NC floats
    __bf16* H1   = (__bf16*)(hagg + NC);        // N*128 bf16
    __bf16* W1pk = H1 + (size_t)N * NHID;       // 8*16*32*16 = 65536 bf16
    __bf16* W2pk = W1pk + 65536;                // 4*4*32*16  = 8192 bf16

    int nb = (N + 255) / 256;
    int eb = (E + 255) / 256;

    // symmetric normalization weights
    init_deg<<<nb, 256, 0, stream>>>(dinv, N);
    accum_deg<<<eb, 256, 0, stream>>>(dst, dinv, E);
    finish_dinv<<<nb, 256, 0, stream>>>(dinv, N);

    // pre-pack weights into WMMA B-fragment layout
    pack_B<<<(4096 + 127) / 128, 128, 0, stream>>>(W1, W1pk, NFEAT, NHID);
    pack_B<<<(512 + 127) / 128, 128, 0, stream>>>(W2, W2pk, NHID, NCLS);

    // MLP via WMMA
    gemm1_relu_wmma<<<N / 16, 256, 0, stream>>>(X, W1pk, b1, H1);
    gemm2_wmma<<<N / 16, 128, 0, stream>>>(H1, W2pk, b2, x0, hcur);

    // K = 10 personalized-pagerank steps
    for (int it = 0; it < 10; ++it) {
        hipMemsetAsync(hagg, 0, NC * sizeof(float), stream);
        edge_scatter<<<(E + 7) / 8, 256, 0, stream>>>(src, dst, dinv, hcur, hagg, E);
        combine<<<(int)((NC + 255) / 256), 256, 0, stream>>>(hagg, x0, dinv, hcur, (int)NC);
    }

    log_softmax64<<<(N + 7) / 8, 256, 0, stream>>>(hcur, out, N);
}
